// enc_dec_attention_28570122453040
// MI455X (gfx1250) — matrix operand order repaired and checked by reading, not compiled, not run on hardware
//
#include <hip/hip_runtime.h>
#include <stdint.h>

#define BATCH 8
#define NN 128
#define CC 256
#define INV_SQRT_DIM 0.0625f   // 1/sqrt(256)

// dynamic LDS layout (dwords): [0,16384) A slab / attn ; [16384, 16384+2*5120) B ping-pong
#define LDSA_DW 16384
#define LDSB_STRIDE 20          // padded row stride (dwords) -> conflict-free frag reads
#define LDSB_DW (256 * LDSB_STRIDE)
#define SMEM_BYTES ((LDSA_DW + 2 * LDSB_DW) * 4)

typedef __attribute__((ext_vector_type(16))) __bf16 v16bf;
typedef __attribute__((ext_vector_type(8)))  float  v8f;

union ABFrag { v16bf v; uint32_t u[8]; };

__device__ __forceinline__ uint32_t pack_bf16x2(float lo, float hi) {
    uint32_t a = __float_as_uint(lo), b = __float_as_uint(hi);
    uint32_t ra = (a + 0x7FFFu + ((a >> 16) & 1u)) >> 16;
    uint32_t rb = (b + 0x7FFFu + ((b >> 16) & 1u)) >> 16;
    return (ra & 0xFFFFu) | (rb << 16);
}
__device__ __forceinline__ uint16_t f2bf(float x) {
    uint32_t a = __float_as_uint(x);
    return (uint16_t)((a + 0x7FFFu + ((a >> 16) & 1u)) >> 16);
}
__device__ __forceinline__ float bf2f(uint32_t h) {
    return __uint_as_float(h << 16);
}

// [128 x 256] fp32 slab -> bf16 in LDS.  Row-major; dword index XOR-swizzled
// with (row&15)<<2 so A-fragment b128 reads (16 lanes, same k-group,
// consecutive rows) hit 16 distinct 4-bank groups, while 4-dword fragment
// groups stay contiguous and 16B aligned.
__device__ __forceinline__ void slab_to_lds_bf16(const float* __restrict__ src,
                                                 uint32_t* lds) {
    int tid = threadIdx.x;
#pragma unroll
    for (int i = 0; i < 64; ++i) {
        int idx = tid + i * 256;          // 0..16383 dword slots
        int row = idx >> 7;
        int d   = idx & 127;
        float2 f = *(const float2*)(src + row * 256 + d * 2);
        lds[(row << 7) | (d ^ ((row & 15) << 2))] = pack_bf16x2(f.x, f.y);
    }
}

// A-fragment (16x32 bf16): lanes 0-15 = rows, khalf = lane>>4.
// VGPR j<4: K = kh*8+2j ; j>=4: K = 16+kh*8+2(j-4).  Two ds_load_b128.
__device__ __forceinline__ v16bf load_a_frag(const uint32_t* lds, int rt, int ks,
                                             int lane) {
    int row = (rt << 4) + (lane & 15);
    int kh  = (lane >> 4) & 1;
    int sw  = (row & 15) << 2;
    const uint32_t* base = lds + (row << 7);
    uint4 x0 = *(const uint4*)(base + (((ks << 4) + (kh << 2)) ^ sw));
    uint4 x1 = *(const uint4*)(base + (((ks << 4) + 8 + (kh << 2)) ^ sw));
    ABFrag f;
    f.u[0] = x0.x; f.u[1] = x0.y; f.u[2] = x0.z; f.u[3] = x0.w;
    f.u[4] = x1.x; f.u[5] = x1.y; f.u[6] = x1.z; f.u[7] = x1.w;
    return f.v;
}

// B-fragment (32x16 bf16) from the LDS stage buffer: lane = column n, kh = lane>>4.
// VGPR j<4: K = kh*8+2j ; j>=4: K = 16+kh*8+2(j-4) (matches A-fragment).
// Row stride 20 dwords => 16 lanes map to 16 distinct banks.
__device__ __forceinline__ v16bf load_b_frag_lds(const uint32_t* bbase, int ct,
                                                 int lane) {
    int n  = (ct << 4) + (lane & 15);
    int kh = (lane >> 4) & 1;
    const uint4* p = (const uint4*)(bbase + n * LDSB_STRIDE + (kh << 2));
    uint4 x0 = p[0];
    uint4 x1 = p[2];
    ABFrag f;
    f.u[0] = x0.x; f.u[1] = x0.y; f.u[2] = x0.z; f.u[3] = x0.w;
    f.u[4] = x1.x; f.u[5] = x1.y; f.u[6] = x1.z; f.u[7] = x1.w;
    return f.v;
}

// Cooperative staged GEMM (all 256 threads): A (swizzled bf16) already in
// lds_a; weight WT (N x K bf16, transposed) streamed through a ping-pong LDS
// stage.  Next K-slice is prefetched into registers while the current slice's
// 16 WMMAs run.  A sched_barrier pins the next B fragment's ds_loads ahead of
// the current WMMA so the allocator must double-buffer them and each
// fragment's LDS latency hides under the preceding WMMA.
__device__ __forceinline__ void block_gemm(const uint32_t* lds_a,
                                           uint32_t* lds_b,
                                           const uint32_t* __restrict__ wt,
                                           int tid, v8f* acc) {
    int lane = tid & 31, wave = tid >> 5;
#pragma unroll
    for (int ct = 0; ct < 16; ++ct) {
        v8f z = {0.f, 0.f, 0.f, 0.f, 0.f, 0.f, 0.f, 0.f};
        acc[ct] = z;
    }
    // stage K-slice 0 (thread t owns WT row n=t: 16 dwords = 4 x b128)
    uint4 st[4];
    {
        const uint4* wr = (const uint4*)(wt + (tid << 7));
#pragma unroll
        for (int p = 0; p < 4; ++p) st[p] = wr[p];
        uint4* bo = (uint4*)(lds_b + tid * LDSB_STRIDE);
#pragma unroll
        for (int p = 0; p < 4; ++p) bo[p] = st[p];
    }
    __syncthreads();
    for (int ks = 0; ks < 8; ++ks) {
        int buf = ks & 1;
        if (ks < 7) {   // prefetch next K-slice into registers (overlaps WMMAs)
            const uint4* wr = (const uint4*)(wt + (tid << 7) + ((ks + 1) << 4));
#pragma unroll
            for (int p = 0; p < 4; ++p) st[p] = wr[p];
        }
        const uint32_t* bbase = lds_b + buf * LDSB_DW;
        v16bf a = load_a_frag(lds_a, wave, ks, lane);
        v16bf b = load_b_frag_lds(bbase, 0, lane);
#pragma unroll
        for (int ct = 0; ct < 16; ++ct) {
            v16bf bn = b;
            if (ct < 15) bn = load_b_frag_lds(bbase, ct + 1, lane);
            // Pin order: next-tile ds_loads must stay ahead of this WMMA so
            // they remain outstanding (wait dscnt<=2) while it executes.
            __builtin_amdgcn_sched_barrier(0);
            acc[ct] = __builtin_amdgcn_wmma_f32_16x16x32_bf16(
                false, a, false, b, (short)0, acc[ct], false, false);
            b = bn;
        }
        __syncthreads();
        if (ks < 7) {
            uint4* bo = (uint4*)(lds_b + (buf ^ 1) * LDSB_DW + tid * LDSB_STRIDE);
#pragma unroll
            for (int p = 0; p < 4; ++p) bo[p] = st[p];
            __syncthreads();
        }
    }
}

// ---------------------------------------------------------------------------
// Weight transpose+convert: WT[n*128 + d] packs bf16(W[2d, n]), bf16(W[2d+1, n])
__global__ __launch_bounds__(256) void wconv_kernel(const float* __restrict__ W,
                                                    uint32_t* __restrict__ WT) {
    int i = blockIdx.x * 256 + threadIdx.x;   // 0..32767
    int n = i >> 7;
    int d = i & 127;
    WT[i] = pack_bf16x2(W[(2 * d) * 256 + n], W[(2 * d + 1) * 256 + n]);
}

// ---------------------------------------------------------------------------
// Generic [128x256] @ [256x256] + bias per batch (q, k, v and node out)
__global__ __launch_bounds__(256) void annot_gemm_kernel(
    const float* __restrict__ A, const uint32_t* __restrict__ WT,
    const float* __restrict__ bias, float* __restrict__ Out) {
    extern __shared__ uint32_t smem[];
    uint32_t* lds_a = smem;
    uint32_t* lds_b = smem + LDSA_DW;
    int b = blockIdx.x;
    int tid = threadIdx.x, lane = tid & 31, wave = tid >> 5;
    slab_to_lds_bf16(A + (size_t)b * NN * CC, lds_a);
    __syncthreads();
    v8f acc[16];
    block_gemm(lds_a, lds_b, WT, tid, acc);
    float* outp = Out + (size_t)b * NN * CC;
#pragma unroll
    for (int ct = 0; ct < 16; ++ct) {
        int col = (ct << 4) + (lane & 15);
        float bs = bias[col];
#pragma unroll
        for (int r = 0; r < 8; ++r) {
            int row = (wave << 4) + r + ((lane >> 4) << 3);
            outp[(size_t)row * CC + col] = acc[ct][r] + bs;
        }
    }
}

// ---------------------------------------------------------------------------
// Fused main kernel: one block per (b, m).
__global__ __launch_bounds__(256) void fused_attn_kernel(
    const float* __restrict__ mol_adj, const float* __restrict__ prot_adj,
    const uint32_t* __restrict__ WvaT, const uint32_t* __restrict__ WkaT,
    const uint32_t* __restrict__ WoedT, const float* __restrict__ bva,
    const float* __restrict__ bka, const float* __restrict__ boed,
    const float* __restrict__ q_ws, const float* __restrict__ k_ws,
    const float* __restrict__ v_ws, float* __restrict__ node_ws,
    float* __restrict__ out_adj) {
    extern __shared__ uint32_t smem[];
    uint32_t* lds_a = smem;                 // A slabs, then attn (bf16)
    uint32_t* lds_b = smem + LDSA_DW;       // weight stage ping-pong
    uint16_t* lds_a16 = (uint16_t*)lds_a;

    int m = blockIdx.x, b = blockIdx.y;
    int tid = threadIdx.x, lane = tid & 31, wave = tid >> 5;
    const size_t slab = ((size_t)(b * NN + m)) * NN * CC;

    // ---- GEMM1: mol_e = mol_adj[b,m] @ Wva + bva  (kept as bf16 in regs) ----
    slab_to_lds_bf16(mol_adj + slab, lds_a);
    // pull the prot_adj slab toward this WGP while GEMM1 runs (128B lines)
    {
        const char* pp = (const char*)(prot_adj + slab);
#pragma unroll
        for (int i = 0; i < 4; ++i)
            __builtin_prefetch(pp + (tid + i * 256) * 128, 0, 1);
    }
    __syncthreads();
    v8f acc[16];
    block_gemm(lds_a, lds_b, WvaT, tid, acc);
    uint32_t mole[16][4];
#pragma unroll
    for (int ct = 0; ct < 16; ++ct) {
        int col = (ct << 4) + (lane & 15);
        float bs = bva[col];
#pragma unroll
        for (int j = 0; j < 4; ++j)
            mole[ct][j] = pack_bf16x2(acc[ct][2 * j] + bs, acc[ct][2 * j + 1] + bs);
    }
    __syncthreads();   // GEMM1 reads of lds_a complete

    // ---- GEMM2: prot_e = prot_adj[b,m] @ Wka + bka (in accumulators) ----
    slab_to_lds_bf16(prot_adj + slab, lds_a);
    __syncthreads();
    block_gemm(lds_a, lds_b, WkaT, tid, acc);
    // (block_gemm ends with a barrier: all lds_a reads complete)

    // ---- elementwise: attn = q*k/16 * (prot_e+1) * mol_e  -> lds_a (bf16) ----
    const float* qrow = q_ws + ((size_t)(b * NN + m)) * CC;
#pragma unroll
    for (int ct = 0; ct < 16; ++ct) {
        int col = (ct << 4) + (lane & 15);
        float bs = bka[col];
        float qv = qrow[col] * INV_SQRT_DIM;
#pragma unroll
        for (int r = 0; r < 8; ++r) {
            int row = (wave << 4) + r + ((lane >> 4) << 3);
            float pe = acc[ct][r] + bs;
            uint32_t w = mole[ct][r >> 1];
            float me = bf2f((r & 1) ? (w >> 16) : (w & 0xFFFFu));
            float kv = k_ws[((size_t)(b * NN + row)) * CC + col];
            float attn = qv * kv * (pe + 1.0f) * me;
            int ds = (col >> 1) ^ ((row & 15) << 2);
            lds_a16[(row << 8) + (ds << 1) + (col & 1)] = f2bf(attn);
        }
    }
    __syncthreads();

    // ---- GEMM3: mol_adj_out = attn @ Woed + boed -> d_out ----
    block_gemm(lds_a, lds_b, WoedT, tid, acc);
    float* outp = out_adj + slab;
#pragma unroll
    for (int ct = 0; ct < 16; ++ct) {
        int col = (ct << 4) + (lane & 15);
        float bs = boed[col];
#pragma unroll
        for (int r = 0; r < 8; ++r) {
            int row = (wave << 4) + r + ((lane >> 4) << 3);
            outp[(size_t)row * CC + col] = acc[ct][r] + bs;
        }
    }

    // ---- softmax over rows (key axis) per column + weighted v sum ----
    {
        int c = tid;   // 256 threads == 256 columns
        float mx = -3.4e38f;
        for (int n = 0; n < NN; ++n) {
            uint32_t h =
                lds_a16[(n << 8) + ((((c >> 1) ^ ((n & 15) << 2)) << 1) | (c & 1))];
            mx = fmaxf(mx, bf2f(h));
        }
        float sum = 0.f, nv = 0.f;
        for (int n = 0; n < NN; ++n) {
            uint32_t h =
                lds_a16[(n << 8) + ((((c >> 1) ^ ((n & 15) << 2)) << 1) | (c & 1))];
            float e = __expf(bf2f(h) - mx);
            sum += e;
            nv += e * v_ws[((size_t)(b * NN + n)) * CC + c];
        }
        node_ws[((size_t)(b * NN + m)) * CC + c] = nv / sum;
    }
}

// ---------------------------------------------------------------------------
__global__ __launch_bounds__(256) void copy4_kernel(const float4* __restrict__ src,
                                                    float4* __restrict__ dst,
                                                    int n4) {
    int i = blockIdx.x * 256 + threadIdx.x;
    if (i < n4) dst[i] = src[i];
}

// ---------------------------------------------------------------------------
extern "C" void kernel_launch(void* const* d_in, const int* in_sizes, int n_in,
                              void* d_out, int out_size, void* d_ws, size_t ws_size,
                              hipStream_t stream) {
    const float* mol_annot  = (const float*)d_in[0];
    const float* prot_annot = (const float*)d_in[1];
    const float* mol_adj    = (const float*)d_in[2];
    const float* prot_adj   = (const float*)d_in[3];
    const float* Wq  = (const float*)d_in[4];  const float* bq  = (const float*)d_in[5];
    const float* Wk  = (const float*)d_in[6];  const float* bk  = (const float*)d_in[7];
    const float* Wv  = (const float*)d_in[8];  const float* bv  = (const float*)d_in[9];
    const float* Wka = (const float*)d_in[10]; const float* bka = (const float*)d_in[11];
    const float* Wva = (const float*)d_in[12]; const float* bva = (const float*)d_in[13];
    const float* Wond= (const float*)d_in[14]; const float* bond= (const float*)d_in[15];
    const float* Woed= (const float*)d_in[16]; const float* boed= (const float*)d_in[17];

    // workspace: 7 bf16 transposed weights (32768 dwords each), then
    // q, k, v, node fp32 buffers of 8*128*256 floats each.
    uint32_t* WqT   = (uint32_t*)d_ws;
    uint32_t* WkT   = WqT   + 32768;
    uint32_t* WvT   = WkT   + 32768;
    uint32_t* WkaT  = WvT   + 32768;
    uint32_t* WvaT  = WkaT  + 32768;
    uint32_t* WoedT = WvaT  + 32768;
    uint32_t* WondT = WoedT + 32768;
    float* q_ws    = (float*)(WondT + 32768);
    float* k_ws    = q_ws + BATCH * NN * CC;
    float* v_ws    = k_ws + BATCH * NN * CC;
    float* node_ws = v_ws + BATCH * NN * CC;

    // output: mol_annot_out | prot_annot | mol_adj_out | prot_adj
    float* out_node   = (float*)d_out;
    float* out_pannot = out_node + BATCH * NN * CC;
    float* out_madj   = out_pannot + BATCH * NN * CC;
    float* out_padj   = out_madj + (size_t)BATCH * NN * NN * CC;

    // 1) weight conversion (bf16, transposed)
    wconv_kernel<<<128, 256, 0, stream>>>(Wq,   WqT);
    wconv_kernel<<<128, 256, 0, stream>>>(Wk,   WkT);
    wconv_kernel<<<128, 256, 0, stream>>>(Wv,   WvT);
    wconv_kernel<<<128, 256, 0, stream>>>(Wka,  WkaT);
    wconv_kernel<<<128, 256, 0, stream>>>(Wva,  WvaT);
    wconv_kernel<<<128, 256, 0, stream>>>(Woed, WoedT);
    wconv_kernel<<<128, 256, 0, stream>>>(Wond, WondT);

    // 2) q, k, v projections
    annot_gemm_kernel<<<BATCH, 256, SMEM_BYTES, stream>>>(mol_annot,  WqT, bq, q_ws);
    annot_gemm_kernel<<<BATCH, 256, SMEM_BYTES, stream>>>(prot_annot, WkT, bk, k_ws);
    annot_gemm_kernel<<<BATCH, 256, SMEM_BYTES, stream>>>(mol_annot,  WvT, bv, v_ws);

    // 3) passthrough outputs (independent of compute chain)
    copy4_kernel<<<(BATCH * NN * CC / 4 + 255) / 256, 256, 0, stream>>>(
        (const float4*)prot_annot, (float4*)out_pannot, BATCH * NN * CC / 4);
    copy4_kernel<<<((int)((size_t)BATCH * NN * NN * CC / 4) + 255) / 256, 256, 0,
                   stream>>>((const float4*)prot_adj, (float4*)out_padj,
                             (int)((size_t)BATCH * NN * NN * CC / 4));

    // 4) fused edge-GEMMs + attention + softmax + node aggregation
    fused_attn_kernel<<<dim3(NN, BATCH), 256, SMEM_BYTES, stream>>>(
        mol_adj, prot_adj, WvaT, WkaT, WoedT, bva, bka, boed, q_ws, k_ws, v_ws,
        node_ws, out_madj);

    // 5) node output projection
    annot_gemm_kernel<<<BATCH, 256, SMEM_BYTES, stream>>>(node_ws, WondT, bond,
                                                          out_node);
}
